// DecoderLayer_63625645523038
// MI455X (gfx1250) — compile-verified, matrix operand order checked by reading, not run on hardware
//
#include <hip/hip_runtime.h>
#include <stdint.h>

// ---------------------------------------------------------------------------
// Types for CDNA5 WMMA (wave32): v16bf A/B operands, v8f accumulator.
// ---------------------------------------------------------------------------
typedef __attribute__((ext_vector_type(16))) __bf16 bf16x16;
typedef __attribute__((ext_vector_type(8)))  float  f32x8;

union Frag { bf16x16 v; uint4 q[2]; };

__device__ __forceinline__ uint16_t f2bf(float f) {
  union { float f; uint32_t u; } x; x.f = f;
  uint32_t r = x.u + 0x7FFFu + ((x.u >> 16) & 1u);   // round-to-nearest-even
  return (uint16_t)(r >> 16);
}

// ---------------------------------------------------------------------------
// CDNA5 async global->LDS copy (no VGPR staging, tracked by ASYNCcnt).
// LDS address = low 32 bits of generic shared pointer (wave-relative offset).
// ---------------------------------------------------------------------------
__device__ __forceinline__ void async_load_b128(void* lds_dst, const void* gsrc) {
  asm volatile("global_load_async_to_lds_b128 %0, %1, off"
               :: "v"((uint32_t)(uintptr_t)lds_dst),
                  "v"((uint64_t)(uintptr_t)gsrc)
               : "memory");
}
__device__ __forceinline__ void wait_async0() {
  asm volatile("s_wait_asynccnt 0x0" ::: "memory");
}

// ---------------------------------------------------------------------------
// Elementwise f32 -> bf16 cast
// ---------------------------------------------------------------------------
__global__ void cast_bf16_k(const float* __restrict__ in, uint16_t* __restrict__ out, long n) {
  long i = (long)blockIdx.x * blockDim.x + threadIdx.x;
  const long stride = (long)gridDim.x * blockDim.x;
  for (; i < n; i += stride) out[i] = f2bf(in[i]);
}

// Transpose + cast: out[c*R + r] = bf16(in[r*C + c])   (in is [R,C] row-major)
__global__ void tcast_bf16_k(const float* __restrict__ in, uint16_t* __restrict__ out, int R, int C) {
  long i = (long)blockIdx.x * blockDim.x + threadIdx.x;
  const long n = (long)R * C;
  const long stride = (long)gridDim.x * blockDim.x;
  for (; i < n; i += stride) {
    const long r = i / C, c = i % C;
    out[c * (long)R + r] = f2bf(in[i]);
  }
}

// bias' = Whead(f32)[N,K] * b_in(f32)[K] + b_head[N]
__global__ void bias_combine_k(const float* __restrict__ Wh, const float* __restrict__ b_in,
                               const float* __restrict__ b_head, float* __restrict__ out,
                               int N, int K) {
  const int i = blockIdx.x * blockDim.x + threadIdx.x;
  if (i >= N) return;
  float s = b_head[i];
  for (int k = 0; k < K; ++k) s += Wh[(size_t)i * K + k] * b_in[k];
  out[i] = s;
}

// ---------------------------------------------------------------------------
// bf16 WMMA GEMM: C[M,N] = A[M,K] * W[N,K]^T (+bias) (+relu) (+f32 residual)
// 128x128x32 block tiles, 8 waves, double-buffered LDS filled by
// global_load_async_to_lds_b128; WMMA compute hides the async latency.
// ---------------------------------------------------------------------------
template<bool RELU, bool RESID, bool OUT_BF, bool OUT_F32>
__global__ __launch_bounds__(256) void gemm_bf16_wmma(
    const uint16_t* __restrict__ A, const uint16_t* __restrict__ W,
    const float* __restrict__ bias, const float* __restrict__ resid,
    uint16_t* __restrict__ outBf, float* __restrict__ outF,
    int M, int N, int K)
{
  constexpr int BM = 128, BN = 128, BK = 32;
  __shared__ __align__(16) uint16_t As[2][BM * BK];
  __shared__ __align__(16) uint16_t Bs[2][BN * BK];

  const int tid  = threadIdx.x;
  const int wave = tid >> 5;
  const int lane = tid & 31;
  const int bm = blockIdx.y * BM;
  const int bn = blockIdx.x * BN;
  const int wm = (wave >> 1) * 32;   // 4 wave-rows of 32
  const int wn = (wave & 1) * 64;    // 2 wave-cols of 64
  // 16-bit A/B fragment lane decode (ISA 7.12.2): lanes<16 -> K base 0, else 8
  const int frow = lane & 15;
  const int fk   = (lane < 16) ? 0 : 8;
  const int rhalf = (lane >> 4) * 8; // C layout: lanes>=16 hold rows M+8

  f32x8 acc[2][4] = {};

  // Issue async copies for one 128x32 A tile + 128x32 B tile (each thread: 4).
  auto issueTiles = [&](int kb, int buf) {
    const int k0 = kb * BK;
#pragma unroll
    for (int i = 0; i < 2; ++i) {
      const int c = tid + i * 256;       // 512 x 16B chunks per buffer pair
      const int r = c >> 2;              // row 0..127
      const int col = (c & 3) * 8;       // half-col 0,8,16,24
      async_load_b128(&As[buf][r * BK + col], &A[(size_t)(bm + r) * K + k0 + col]);
      async_load_b128(&Bs[buf][r * BK + col], &W[(size_t)(bn + r) * K + k0 + col]);
    }
  };

  const int nk = K / BK;
  issueTiles(0, 0);
  wait_async0();
  __syncthreads();

  for (int kb = 0; kb < nk; ++kb) {
    const int cur = kb & 1;
    if (kb + 1 < nk) issueTiles(kb + 1, cur ^ 1);   // async fill of next buffer
    if (kb + 4 < nk) {   // L2 prefetch (global_prefetch_b8)
      __builtin_prefetch(&A[(size_t)(bm + (tid >> 1)) * K + (size_t)(kb + 4) * BK], 0, 1);
      __builtin_prefetch(&W[(size_t)(bn + (tid >> 1)) * K + (size_t)(kb + 4) * BK], 0, 1);
    }

    Frag a[2], b[4];
#pragma unroll
    for (int i = 0; i < 2; ++i) {
      const uint16_t* p = &As[cur][(wm + i * 16 + frow) * BK + fk];
      a[i].q[0] = *reinterpret_cast<const uint4*>(p);
      a[i].q[1] = *reinterpret_cast<const uint4*>(p + 16);
    }
#pragma unroll
    for (int j = 0; j < 4; ++j) {
      const uint16_t* p = &Bs[cur][(wn + j * 16 + frow) * BK + fk];
      b[j].q[0] = *reinterpret_cast<const uint4*>(p);
      b[j].q[1] = *reinterpret_cast<const uint4*>(p + 16);
    }
#pragma unroll
    for (int i = 0; i < 2; ++i)
#pragma unroll
      for (int j = 0; j < 4; ++j)
        acc[i][j] = __builtin_amdgcn_wmma_f32_16x16x32_bf16(
            false, a[i].v, false, b[j].v, (short)0, acc[i][j], false, false);

    wait_async0();      // next buffer resident (own wave's copies)
    __syncthreads();    // all waves' copies resident; cur reads complete
  }

#pragma unroll
  for (int i = 0; i < 2; ++i) {
#pragma unroll
    for (int j = 0; j < 4; ++j) {
      const int gn = bn + wn + j * 16 + frow;
      const float bv = bias ? bias[gn] : 0.0f;
#pragma unroll
      for (int v = 0; v < 8; ++v) {
        const int gm = bm + wm + i * 16 + rhalf + v;
        float val = acc[i][j][v] + bv;
        if (RELU)  val = fmaxf(val, 0.0f);
        if (RESID) val += resid[(size_t)gm * N + gn];
        if (OUT_F32) outF[(size_t)gm * N + gn]  = val;
        if (OUT_BF)  outBf[(size_t)gm * N + gn] = f2bf(val);
      }
    }
  }
}

// ---------------------------------------------------------------------------
// Flash attention (bf16 WMMA, f32 online softmax). 4 waves, 64-row Q blocks,
// 64-key iterations, DK=DV=128, heads packed at stride 1024.
// Q/K tiles: async global->LDS. V tile: batched b128 loads + LDS transpose.
// ---------------------------------------------------------------------------
template<bool CAUSAL>
__global__ __launch_bounds__(128) void attn_wmma(
    const uint16_t* __restrict__ Qh, const uint16_t* __restrict__ Kh,
    const uint16_t* __restrict__ Vh, const int* __restrict__ pad,
    uint16_t* __restrict__ Out, int Sq, int Sk)
{
  constexpr int DK_ = 128, BQ = 64, BKV = 64, LD = 1024;
  const int b = blockIdx.x >> 3;
  const int h = blockIdx.x & 7;
  const int q0 = blockIdx.y * BQ;
  const int wave = threadIdx.x >> 5;
  const int lane = threadIdx.x & 31;
  const int frow = lane & 15;
  const int fk = (lane < 16) ? 0 : 8;
  const int rhalf = (lane >> 4) * 8;

  __shared__ __align__(16) uint16_t Qs[BQ * DK_];       // 16 KB
  __shared__ __align__(16) uint16_t Ks[BKV * DK_];      // 16 KB
  __shared__ __align__(16) uint16_t Vt[DK_ * BKV];      // 16 KB, transposed [dv][key]
  __shared__ __align__(16) uint16_t Ps[4][16 * BKV];    // 8 KB, per-wave P tiles

  const uint16_t* Qg = Qh + (size_t)b * Sq * LD + h * DK_;
  const uint16_t* Kg = Kh + (size_t)b * Sk * LD + h * DK_;
  const uint16_t* Vg = Vh + (size_t)b * Sk * LD + h * DK_;

  // async copy of the Q tile (reused across the whole key loop)
  for (int c = threadIdx.x; c < BQ * DK_ / 8; c += 128) {
    const int r = c >> 4, col = (c & 15) * 8;
    async_load_b128(&Qs[r * DK_ + col], &Qg[(size_t)(q0 + r) * LD + col]);
  }

  f32x8 o[8] = {};
  float rmax[8], rsum[8];
#pragma unroll
  for (int v = 0; v < 8; ++v) { rmax[v] = -1e30f; rsum[v] = 0.0f; }
  wait_async0();
  __syncthreads();

  const int kend = CAUSAL ? (q0 + BQ) : Sk;
  const float scale = 0.088388347648318447f;  // 1/sqrt(128)

  for (int k0 = 0; k0 < kend; k0 += BKV) {
    // K tile: async global->LDS (natural [key][dk] layout feeds B-fragments)
#pragma unroll
    for (int it = 0; it < 8; ++it) {
      const int c = threadIdx.x + it * 128;
      const int r = c >> 4, col = (c & 15) * 8;
      async_load_b128(&Ks[r * DK_ + col], &Kg[(size_t)(k0 + r) * LD + col]);
    }
    // V tile: batch all 8 global b128 loads, then one wait, then LDS transpose
    uint4 vr[8];
#pragma unroll
    for (int it = 0; it < 8; ++it) {
      const int c = threadIdx.x + it * 128;
      const int r = c >> 4, col = (c & 15) * 8;
      vr[it] = *reinterpret_cast<const uint4*>(&Vg[(size_t)(k0 + r) * LD + col]);
    }
#pragma unroll
    for (int it = 0; it < 8; ++it) {
      const int c = threadIdx.x + it * 128;
      const int r = c >> 4, col = (c & 15) * 8;
      const uint16_t* hv = reinterpret_cast<const uint16_t*>(&vr[it]);
#pragma unroll
      for (int e = 0; e < 8; ++e) Vt[(col + e) * BKV + r] = hv[e];
    }
    wait_async0();
    __syncthreads();

    // S = Q * K^T   (16x64 stripe per wave)
    f32x8 s[4] = {};
#pragma unroll
    for (int kc = 0; kc < 4; ++kc) {
      Frag aq;
      const uint16_t* p = &Qs[(wave * 16 + frow) * DK_ + kc * 32 + fk];
      aq.q[0] = *reinterpret_cast<const uint4*>(p);
      aq.q[1] = *reinterpret_cast<const uint4*>(p + 16);
#pragma unroll
      for (int j = 0; j < 4; ++j) {
        Frag bk;
        const uint16_t* pb = &Ks[(j * 16 + frow) * DK_ + kc * 32 + fk];
        bk.q[0] = *reinterpret_cast<const uint4*>(pb);
        bk.q[1] = *reinterpret_cast<const uint4*>(pb + 16);
        s[j] = __builtin_amdgcn_wmma_f32_16x16x32_bf16(
            false, aq.v, false, bk.v, (short)0, s[j], false, false);
      }
    }

    // scale + mask + online softmax (stats live in C-matrix layout)
    float nmax[8];
#pragma unroll
    for (int v = 0; v < 8; ++v) nmax[v] = rmax[v];
#pragma unroll
    for (int j = 0; j < 4; ++j) {
      const int kcol = k0 + j * 16 + frow;
      const int ok_pad = pad ? pad[(size_t)b * Sk + kcol] : 1;
#pragma unroll
      for (int v = 0; v < 8; ++v) {
        const int qrow = q0 + wave * 16 + rhalf + v;
        float val = s[j][v] * scale;
        const bool ok = (ok_pad != 0) && (!CAUSAL || (kcol <= qrow));
        val = ok ? val : -1e30f;
        s[j][v] = val;
        nmax[v] = fmaxf(nmax[v], val);
      }
    }
#pragma unroll
    for (int v = 0; v < 8; ++v)
#pragma unroll
      for (int m = 1; m < 16; m <<= 1)   // reduce across N (16 lanes per half)
        nmax[v] = fmaxf(nmax[v], __shfl_xor(nmax[v], m, 32));

#pragma unroll
    for (int v = 0; v < 8; ++v) {
      const float corr = __expf(rmax[v] - nmax[v]);
      rmax[v] = nmax[v];
      rsum[v] *= corr;
#pragma unroll
      for (int t = 0; t < 8; ++t) o[t][v] *= corr;
    }

    // exp, accumulate partial row-sums, round-trip P through LDS (C->A layout)
#pragma unroll
    for (int j = 0; j < 4; ++j)
#pragma unroll
      for (int v = 0; v < 8; ++v) {
        const float pexp = __expf(s[j][v] - rmax[v]);
        rsum[v] += pexp;
        Ps[wave][(rhalf + v) * BKV + j * 16 + frow] = f2bf(pexp);
      }
    __syncthreads();

    // O += P(16x64) * V(64x128); V fragments from transposed LDS tile
#pragma unroll
    for (int kc = 0; kc < 2; ++kc) {
      Frag ap;
      const uint16_t* p = &Ps[wave][frow * BKV + kc * 32 + fk];
      ap.q[0] = *reinterpret_cast<const uint4*>(p);
      ap.q[1] = *reinterpret_cast<const uint4*>(p + 16);
#pragma unroll
      for (int t = 0; t < 8; ++t) {
        Frag bv;
        const uint16_t* pb = &Vt[(t * 16 + frow) * BKV + kc * 32 + fk];
        bv.q[0] = *reinterpret_cast<const uint4*>(pb);
        bv.q[1] = *reinterpret_cast<const uint4*>(pb + 16);
        o[t] = __builtin_amdgcn_wmma_f32_16x16x32_bf16(
            false, ap.v, false, bv.v, (short)0, o[t], false, false);
      }
    }
    __syncthreads();
  }

#pragma unroll
  for (int v = 0; v < 8; ++v)
#pragma unroll
    for (int m = 1; m < 16; m <<= 1)
      rsum[v] += __shfl_xor(rsum[v], m, 32);

  uint16_t* Og = Out + (size_t)b * Sq * LD + h * DK_;
#pragma unroll
  for (int t = 0; t < 8; ++t) {
    const int n = t * 16 + frow;
#pragma unroll
    for (int v = 0; v < 8; ++v) {
      const int r = q0 + wave * 16 + rhalf + v;
      Og[(size_t)r * LD + n] = f2bf(o[t][v] / fmaxf(rsum[v], 1e-30f));
    }
  }
}

// ---------------------------------------------------------------------------
// LayerNorm over rows of D=1024; optional f32 and bf16 outputs.
// ---------------------------------------------------------------------------
__global__ __launch_bounds__(256) void layernorm_k(
    const float* __restrict__ in, const float* __restrict__ g, const float* __restrict__ bta,
    float* __restrict__ outF, uint16_t* __restrict__ outBf, int D)
{
  const int row = blockIdx.x;
  const float* x = in + (size_t)row * D;
  float s = 0.f, s2 = 0.f;
  for (int i = threadIdx.x; i < D; i += 256) { const float v = x[i]; s += v; s2 += v * v; }
#pragma unroll
  for (int m = 16; m >= 1; m >>= 1) { s += __shfl_xor(s, m, 32); s2 += __shfl_xor(s2, m, 32); }
  __shared__ float rs[8], rs2[8];
  if ((threadIdx.x & 31) == 0) { rs[threadIdx.x >> 5] = s; rs2[threadIdx.x >> 5] = s2; }
  __syncthreads();
  s = 0.f; s2 = 0.f;
#pragma unroll
  for (int w = 0; w < 8; ++w) { s += rs[w]; s2 += rs2[w]; }
  const float mu = s / D;
  const float var = fmaxf(s2 / D - mu * mu, 0.f);
  const float rstd = rsqrtf(var + 1e-5f);
  for (int i = threadIdx.x; i < D; i += 256) {
    const float v = (x[i] - mu) * rstd * g[i] + bta[i];
    if (outF)  outF[(size_t)row * D + i]  = v;
    if (outBf) outBf[(size_t)row * D + i] = f2bf(v);
  }
}

// ---------------------------------------------------------------------------
// Host-side orchestration
// ---------------------------------------------------------------------------
extern "C" void kernel_launch(void* const* d_in, const int* in_sizes, int n_in,
                              void* d_out, int out_size, void* d_ws, size_t ws_size,
                              hipStream_t stream)
{
  (void)in_sizes; (void)n_in; (void)out_size; (void)ws_size;
  constexpr int B = 4, SD = 2048, SE = 2048, DM = 1024, H = 8, DFF = 4096;
  constexpr int Mq = B * SD;  // 8192
  constexpr int Me = B * SE;  // 8192

  const float* x_in   = (const float*)d_in[0];
  const float* enc    = (const float*)d_in[1];
  const int*   inpad  = (const int*)d_in[2];
  const int*   outpad = (const int*)d_in[3];
  const float* w_qkv[6]; const float* b_qkv[6];
  for (int i = 0; i < 6; ++i) { w_qkv[i] = (const float*)d_in[4 + 2 * i]; b_qkv[i] = (const float*)d_in[5 + 2 * i]; }
  const float* sa_ow = (const float*)d_in[22], *sa_ob = (const float*)d_in[23];
  const float* ed_ow = (const float*)d_in[30], *ed_ob = (const float*)d_in[31];
  const float* ff_w1 = (const float*)d_in[32], *ff_b1 = (const float*)d_in[33];
  const float* ff_w2 = (const float*)d_in[34], *ff_b2 = (const float*)d_in[35];
  const float* ln1g = (const float*)d_in[36], *ln1b = (const float*)d_in[37];
  const float* ln2g = (const float*)d_in[38], *ln2b = (const float*)d_in[39];
  const float* headW[6] = { (const float*)d_in[16], (const float*)d_in[18], (const float*)d_in[20],
                            (const float*)d_in[24], (const float*)d_in[26], (const float*)d_in[28] };
  const float* headB[6] = { (const float*)d_in[17], (const float*)d_in[19], (const float*)d_in[21],
                            (const float*)d_in[25], (const float*)d_in[27], (const float*)d_in[29] };

  char* base = (char*)d_ws;
  size_t off = 0;
  auto alloc = [&](size_t bytes) -> char* {
    char* p = base + off;
    off = (off + bytes + 255) & ~(size_t)255;
    return p;
  };
  auto abf  = [&](size_t e) -> uint16_t* { return (uint16_t*)alloc(e * 2); };
  auto af32 = [&](size_t e) -> float*    { return (float*)alloc(e * 4); };

  uint16_t* Xd = abf((size_t)Mq * DM);
  uint16_t* Xe = abf((size_t)Me * DM);
  uint16_t* Whead[6]; for (int i = 0; i < 6; ++i) Whead[i] = abf((size_t)DM * DM);
  uint16_t* Wsao = abf((size_t)DM * DM);
  uint16_t* Wedo = abf((size_t)DM * DM);
  uint16_t* Wff1 = abf((size_t)DFF * DM);
  uint16_t* Wff2 = abf((size_t)DM * DFF);
  uint16_t* Wt   = abf((size_t)DM * DM);
  uint16_t* Wc[6]; for (int i = 0; i < 6; ++i) Wc[i] = abf((size_t)DM * DM);
  float*    bc[6]; for (int i = 0; i < 6; ++i) bc[i] = af32(DM);
  uint16_t* Qh = abf((size_t)Mq * DM);
  uint16_t* Kh = abf((size_t)Me * DM);
  uint16_t* Vh = abf((size_t)Me * DM);
  uint16_t* AO = abf((size_t)Mq * DM);
  float*    T0 = af32((size_t)Mq * DM);
  uint16_t* Xbf = abf((size_t)Mq * DM);
  float*    Yf  = af32((size_t)Mq * DM);
  uint16_t* Ybf = abf((size_t)Mq * DM);
  uint16_t* FFh = abf((size_t)Mq * DFF);

  auto cast = [&](const float* s, uint16_t* d, size_t n) {
    cast_bf16_k<<<2048, 256, 0, stream>>>(s, d, (long)n);
  };
  cast(x_in, Xd, (size_t)Mq * DM);
  cast(enc,  Xe, (size_t)Me * DM);
  for (int i = 0; i < 6; ++i) cast(headW[i], Whead[i], (size_t)DM * DM);
  cast(sa_ow, Wsao, (size_t)DM * DM);
  cast(ed_ow, Wedo, (size_t)DM * DM);
  cast(ff_w1, Wff1, (size_t)DFF * DM);
  cast(ff_w2, Wff2, (size_t)DM * DFF);

  // Fold per-head projection into the preceding Linear:
  //   Wc = Whead * W1  (so Qh = X * Wc^T + (Whead*b1 + bhead))
  for (int i = 0; i < 6; ++i) {
    tcast_bf16_k<<<2048, 256, 0, stream>>>(w_qkv[i], Wt, DM, DM);     // Wt[d][o] = W1[o][d]
    dim3 g(DM / 128, DM / 128);
    gemm_bf16_wmma<false, false, true, false><<<g, 256, 0, stream>>>(
        Whead[i], Wt, nullptr, nullptr, Wc[i], nullptr, DM, DM, DM);
    bias_combine_k<<<DM / 256, 256, 0, stream>>>(headW[i], b_qkv[i], headB[i], bc[i], DM, DM);
  }

  // ---- self attention ----
  {
    dim3 g(DM / 128, Mq / 128);
    gemm_bf16_wmma<false, false, true, false><<<g, 256, 0, stream>>>(Xd, Wc[0], bc[0], nullptr, Qh, nullptr, Mq, DM, DM);
    gemm_bf16_wmma<false, false, true, false><<<g, 256, 0, stream>>>(Xd, Wc[1], bc[1], nullptr, Kh, nullptr, Mq, DM, DM);
    gemm_bf16_wmma<false, false, true, false><<<g, 256, 0, stream>>>(Xd, Wc[2], bc[2], nullptr, Vh, nullptr, Mq, DM, DM);
  }
  { dim3 ga(B * H, SD / 64);
    attn_wmma<true><<<ga, 128, 0, stream>>>(Qh, Kh, Vh, outpad, AO, SD, SD); }
  { dim3 g(DM / 128, Mq / 128);
    gemm_bf16_wmma<false, true, false, true><<<g, 256, 0, stream>>>(AO, Wsao, sa_ob, x_in, nullptr, T0, Mq, DM, DM); }
  layernorm_k<<<Mq, 256, 0, stream>>>(T0, ln1g, ln1b, nullptr, Xbf, DM);

  // ---- cross attention ----
  {
    dim3 g(DM / 128, Mq / 128);
    gemm_bf16_wmma<false, false, true, false><<<g, 256, 0, stream>>>(Xbf, Wc[3], bc[3], nullptr, Qh, nullptr, Mq, DM, DM);
    gemm_bf16_wmma<false, false, true, false><<<g, 256, 0, stream>>>(Xe,  Wc[4], bc[4], nullptr, Kh, nullptr, Me, DM, DM);
    gemm_bf16_wmma<false, false, true, false><<<g, 256, 0, stream>>>(Xe,  Wc[5], bc[5], nullptr, Vh, nullptr, Me, DM, DM);
  }
  { dim3 ga(B * H, SD / 64);
    attn_wmma<false><<<ga, 128, 0, stream>>>(Qh, Kh, Vh, inpad, AO, SD, SE); }
  { dim3 g(DM / 128, Mq / 128);
    gemm_bf16_wmma<false, true, false, true><<<g, 256, 0, stream>>>(AO, Wedo, ed_ob, x_in, nullptr, T0, Mq, DM, DM); }
  layernorm_k<<<Mq, 256, 0, stream>>>(T0, ln2g, ln2b, Yf, Ybf, DM);

  // ---- FFN ----
  { dim3 g(DFF / 128, Mq / 128);
    gemm_bf16_wmma<true, false, true, false><<<g, 256, 0, stream>>>(Ybf, Wff1, ff_b1, nullptr, FFh, nullptr, Mq, DFF, DM); }
  { dim3 g(DM / 128, Mq / 128);
    gemm_bf16_wmma<false, true, false, true><<<g, 256, 0, stream>>>(FFh, Wff2, ff_b2, Yf, nullptr, T0, Mq, DM, DFF); }
  layernorm_k<<<Mq, 256, 0, stream>>>(T0, ln2g, ln2b, (float*)d_out, nullptr, DM);
}
